// MultiHeadAttention_51823075393649
// MI455X (gfx1250) — matrix operand order repaired and checked by reading, not compiled, not run on hardware
//
#include <hip/hip_runtime.h>
#include <hip/hip_bf16.h>

typedef __bf16 bf16_t;
typedef __attribute__((ext_vector_type(16))) __bf16 v16bf;
typedef __attribute__((ext_vector_type(8)))  __bf16 v8bf;
typedef __attribute__((ext_vector_type(8)))  float  v8f;
typedef __attribute__((ext_vector_type(4)))  float  v4f;

#define D_MODEL 1024
#define SEQ     2048
#define BATCH   2
#define NHEAD   16
#define DKH     64

static __device__ __forceinline__ v8f wmma_bf16(v16bf a, v16bf b, v8f c) {
  // D = A(16x32) * B(32x16) + C, f32 accumulate
  return __builtin_amdgcn_wmma_f32_16x16x32_bf16(false, a, false, b, (short)0, c,
                                                 false, false);
}

// ---------------------------------------------------------------------------
// fp32 -> bf16 weight conversion (done once per weight matrix)
// ---------------------------------------------------------------------------
__global__ __launch_bounds__(256) void cvt_f32_to_bf16(const float* __restrict__ src,
                                                       bf16_t* __restrict__ dst) {
  int i = (blockIdx.x * blockDim.x + threadIdx.x) * 4;
  v4f v = *(const v4f*)(src + i);
  dst[i + 0] = (bf16_t)v[0];
  dst[i + 1] = (bf16_t)v[1];
  dst[i + 2] = (bf16_t)v[2];
  dst[i + 3] = (bf16_t)v[3];
}

// ---------------------------------------------------------------------------
// Projection GEMM: out = X(fp32)[4096,1024] @ W(bf16)[1024,1024]^T + bias
// One wave computes a 16(M) x 64(N) tile. OUT_VT=1 stores V transposed as
// [B,H,Dk,S] so the attention PV GEMM gets contiguous B-operand loads.
// ---------------------------------------------------------------------------
template <int OUT_VT>
__global__ __launch_bounds__(32) void proj_kernel(const float* __restrict__ X,
                                                  const bf16_t* __restrict__ Wb,
                                                  const float* __restrict__ bias,
                                                  bf16_t* __restrict__ out) {
  const int l      = threadIdx.x;
  const int lane16 = l & 15;
  const int hi     = l >> 4;
  const int row0   = blockIdx.x * 16;
  const int col0   = blockIdx.y * 64;

  v8f acc[4];
#pragma unroll
  for (int t = 0; t < 4; ++t) {
    float bv = bias[col0 + t * 16 + lane16];
#pragma unroll
    for (int j = 0; j < 8; ++j) acc[t][j] = bv;
  }

  const float* xrow = X + (size_t)(row0 + lane16) * D_MODEL;

  for (int k0 = 0; k0 < D_MODEL; k0 += 32) {
    // A operand: 16x32 bf16. lane = M row; elems 0..15 -> K=hi*16+0..15
    // (contiguous, matching B).
    v16bf a;
    {
      const float* p0 = xrow + k0 + hi * 16;
      const float* p1 = xrow + k0 + hi * 16 + 8;
      v4f x0 = *(const v4f*)p0;
      v4f x1 = *(const v4f*)(p0 + 4);
      v4f x2 = *(const v4f*)p1;
      v4f x3 = *(const v4f*)(p1 + 4);
#pragma unroll
      for (int j = 0; j < 4; ++j) {
        a[j]      = (bf16_t)x0[j];
        a[4 + j]  = (bf16_t)x1[j];
        a[8 + j]  = (bf16_t)x2[j];
        a[12 + j] = (bf16_t)x3[j];
      }
    }
    __builtin_prefetch(xrow + k0 + 64, 0, 1);

#pragma unroll
    for (int t = 0; t < 4; ++t) {
      // B operand: 32x16 bf16. lane = N col (weight row, since we need W^T);
      // elems 0..15 -> K = hi*16 + 0..15 (contiguous 32B load).
      const bf16_t* wp =
          Wb + (size_t)(col0 + t * 16 + lane16) * D_MODEL + k0 + hi * 16;
      v16bf b = *(const v16bf*)wp;
      acc[t]  = wmma_bf16(a, b, acc[t]);
    }
  }

  // C layout: VGPR v, lane -> out[row0 + v + 8*hi][col0 + t*16 + lane16]
#pragma unroll
  for (int t = 0; t < 4; ++t) {
    int col = col0 + t * 16 + lane16;
#pragma unroll
    for (int v = 0; v < 8; ++v) {
      int   row = row0 + v + 8 * hi;
      float val = acc[t][v];
      if (OUT_VT == 0) {
        out[(size_t)row * D_MODEL + col] = (bf16_t)val;
      } else {
        int b  = row >> 11;          // row / SEQ
        int sk = row & (SEQ - 1);
        int h  = col >> 6;           // col / DKH
        int d  = col & (DKH - 1);
        out[(((size_t)(b * NHEAD + h)) * DKH + d) * SEQ + sk] = (bf16_t)val;
      }
    }
  }
}

// ---------------------------------------------------------------------------
// Two-pass flash attention: one wave per (b, h, 16 query rows).
// Pass 1: stream K, track per-lane score max in registers (no cross-lane ops
//         in the loop); single butterfly reduction at the end.
// Pass 2: with the row max fixed, no O-rescale and no per-iteration sum
//         reduction: exp -> per-lane partial sums; P transposed via a 1 KB
//         LDS bounce into A-operand layout; O += P @ V on the XDL pipe.
// K/V are L2-resident (8 MB each vs 192 MB L2), so the re-read is cheap.
// ---------------------------------------------------------------------------
__global__ __launch_bounds__(32) void attn_kernel(const bf16_t* __restrict__ Qp,
                                                  const bf16_t* __restrict__ Kp,
                                                  const bf16_t* __restrict__ Vt,
                                                  bf16_t* __restrict__ Ob) {
  const int l      = threadIdx.x;
  const int lane16 = l & 15;
  const int hi     = l >> 4;
  const int bh     = blockIdx.x;      // 0..31
  const int b      = bh >> 4;
  const int h      = bh & 15;
  const int sq0    = blockIdx.y * 16; // query row base

  __shared__ __align__(32) bf16_t pbuf[16 * 32];  // P transpose bounce buffer

  // Preload Q tile (16x64) as two A operands (K = 0..31 and 32..63)
  const bf16_t* qrow =
      Qp + ((size_t)(b * SEQ) + sq0 + lane16) * D_MODEL + h * DKH;
  v16bf qa[2];
#pragma unroll
  for (int c = 0; c < 2; ++c) {
    v8bf q0 = *(const v8bf*)(qrow + c * 32 + hi * 16);
    v8bf q1 = *(const v8bf*)(qrow + c * 32 + hi * 16 + 8);
#pragma unroll
    for (int j = 0; j < 8; ++j) {
      qa[c][j]     = q0[j];
      qa[c][8 + j] = q1[j];
    }
  }

  const bf16_t* kbase  = Kp + (size_t)(b * SEQ) * D_MODEL + h * DKH;
  const bf16_t* vbase  = Vt + ((size_t)(b * NHEAD + h)) * DKH * SEQ;
  const float   scale  = 0.125f;  // 1/sqrt(Dk)

  // ---------------- Pass 1: per-row score max ----------------
  float lmax[8];
#pragma unroll
  for (int v = 0; v < 8; ++v) lmax[v] = -1e30f;

  for (int kt = 0; kt < SEQ; kt += 32) {
#pragma unroll
    for (int nt = 0; nt < 2; ++nt) {
      const bf16_t* krow = kbase + (size_t)(kt + nt * 16 + lane16) * D_MODEL;
      v16bf kb0 = *(const v16bf*)(krow + hi * 16);
      v16bf kb1 = *(const v16bf*)(krow + 32 + hi * 16);
      v8f   z;
#pragma unroll
      for (int v = 0; v < 8; ++v) z[v] = 0.0f;
      z = wmma_bf16(qa[0], kb0, z);
      z = wmma_bf16(qa[1], kb1, z);
#pragma unroll
      for (int v = 0; v < 8; ++v) lmax[v] = fmaxf(lmax[v], z[v]);
    }
  }
  // single cross-lane butterfly (rows replicate over the 16 lanes sharing hi)
#pragma unroll
  for (int m = 1; m < 16; m <<= 1)
#pragma unroll
    for (int v = 0; v < 8; ++v)
      lmax[v] = fmaxf(lmax[v], __shfl_xor(lmax[v], m, 32));

  float ms[8];  // scaled row max
#pragma unroll
  for (int v = 0; v < 8; ++v) ms[v] = lmax[v] * scale;

  // ---------------- Pass 2: exp / PV accumulate ----------------
  float psum[8];
  v8f   oacc[4];
#pragma unroll
  for (int v = 0; v < 8; ++v) psum[v] = 0.0f;
#pragma unroll
  for (int t = 0; t < 4; ++t)
#pragma unroll
    for (int v = 0; v < 8; ++v) oacc[t][v] = 0.0f;

  for (int kt = 0; kt < SEQ; kt += 32) {
    // scores for 32 keys (two 16x16 C tiles)
    v8f sacc[2];
#pragma unroll
    for (int nt = 0; nt < 2; ++nt) {
      const bf16_t* krow = kbase + (size_t)(kt + nt * 16 + lane16) * D_MODEL;
      v16bf kb0 = *(const v16bf*)(krow + hi * 16);
      v16bf kb1 = *(const v16bf*)(krow + 32 + hi * 16);
      v8f   z;
#pragma unroll
      for (int v = 0; v < 8; ++v) z[v] = 0.0f;
      z        = wmma_bf16(qa[0], kb0, z);
      z        = wmma_bf16(qa[1], kb1, z);
      sacc[nt] = z;
    }

    // p = exp(s*scale - m); accumulate per-lane partial sums (no shuffles)
#pragma unroll
    for (int nt = 0; nt < 2; ++nt)
#pragma unroll
      for (int v = 0; v < 8; ++v) {
        float e      = __expf(fmaf(sacc[nt][v], scale, -ms[v]));
        sacc[nt][v]  = e;
        psum[v]     += e;
      }

    // transpose P (C layout) -> A layout via LDS
    __syncthreads();
#pragma unroll
    for (int nt = 0; nt < 2; ++nt)
#pragma unroll
      for (int v = 0; v < 8; ++v)
        pbuf[(v + 8 * hi) * 32 + nt * 16 + lane16] = (bf16_t)sacc[nt][v];
    __syncthreads();

    v16bf pa;
    {
      const bf16_t* pr = pbuf + lane16 * 32;
      v8bf p0 = *(const v8bf*)(pr + hi * 16);
      v8bf p1 = *(const v8bf*)(pr + hi * 16 + 8);
#pragma unroll
      for (int j = 0; j < 8; ++j) {
        pa[j]     = p0[j];
        pa[8 + j] = p1[j];
      }
    }

    // O += P @ V (V pre-transposed [B,H,Dk,S] -> contiguous 32B B loads)
#pragma unroll
    for (int t = 0; t < 4; ++t) {
      const bf16_t* vp =
          vbase + (size_t)(t * 16 + lane16) * SEQ + kt + hi * 16;
      v16bf vb = *(const v16bf*)vp;
      oacc[t]  = wmma_bf16(pa, vb, oacc[t]);
    }
  }

  // single cross-lane sum reduction for the softmax denominator
#pragma unroll
  for (int m = 1; m < 16; m <<= 1)
#pragma unroll
    for (int v = 0; v < 8; ++v) psum[v] += __shfl_xor(psum[v], m, 32);

  // ---- finalize: O / l, heads concatenated -> Ob[B*S, D] ----
  float rinv[8];
#pragma unroll
  for (int v = 0; v < 8; ++v) rinv[v] = 1.0f / psum[v];
#pragma unroll
  for (int t = 0; t < 4; ++t)
#pragma unroll
    for (int v = 0; v < 8; ++v) {
      int   row = sq0 + v + 8 * hi;
      int   col = h * DKH + t * 16 + lane16;
      float o   = oacc[t][v] * rinv[v];
      Ob[((size_t)(b * SEQ) + row) * D_MODEL + col] = (bf16_t)o;
    }
}

// ---------------------------------------------------------------------------
// Output projection: out(fp32) = Ob(bf16)[4096,1024] @ Wo(bf16)^T + b_o
// ---------------------------------------------------------------------------
__global__ __launch_bounds__(32) void out_proj_kernel(const bf16_t* __restrict__ X,
                                                      const bf16_t* __restrict__ Wb,
                                                      const float* __restrict__ bias,
                                                      float* __restrict__ out) {
  const int l      = threadIdx.x;
  const int lane16 = l & 15;
  const int hi     = l >> 4;
  const int row0   = blockIdx.x * 16;
  const int col0   = blockIdx.y * 64;

  v8f acc[4];
#pragma unroll
  for (int t = 0; t < 4; ++t) {
    float bv = bias[col0 + t * 16 + lane16];
#pragma unroll
    for (int j = 0; j < 8; ++j) acc[t][j] = bv;
  }

  const bf16_t* xrow = X + (size_t)(row0 + lane16) * D_MODEL;

  for (int k0 = 0; k0 < D_MODEL; k0 += 32) {
    v16bf a;
    {
      v8bf x0 = *(const v8bf*)(xrow + k0 + hi * 16);
      v8bf x1 = *(const v8bf*)(xrow + k0 + hi * 16 + 8);
#pragma unroll
      for (int j = 0; j < 8; ++j) {
        a[j]     = x0[j];
        a[8 + j] = x1[j];
      }
    }
#pragma unroll
    for (int t = 0; t < 4; ++t) {
      const bf16_t* wp =
          Wb + (size_t)(col0 + t * 16 + lane16) * D_MODEL + k0 + hi * 16;
      v16bf b = *(const v16bf*)wp;
      acc[t]  = wmma_bf16(a, b, acc[t]);
    }
  }

#pragma unroll
  for (int t = 0; t < 4; ++t) {
    int col = col0 + t * 16 + lane16;
#pragma unroll
    for (int v = 0; v < 8; ++v) {
      int row = row0 + v + 8 * hi;
      out[(size_t)row * D_MODEL + col] = acc[t][v];
    }
  }
}

// ---------------------------------------------------------------------------
extern "C" void kernel_launch(void* const* d_in, const int* in_sizes, int n_in,
                              void* d_out, int out_size, void* d_ws, size_t ws_size,
                              hipStream_t stream) {
  const float* q   = (const float*)d_in[0];
  const float* k   = (const float*)d_in[1];
  const float* v   = (const float*)d_in[2];
  const float* w_q = (const float*)d_in[3];
  const float* b_q = (const float*)d_in[4];
  const float* w_k = (const float*)d_in[5];
  const float* b_k = (const float*)d_in[6];
  const float* w_v = (const float*)d_in[7];
  const float* b_v = (const float*)d_in[8];
  const float* w_o = (const float*)d_in[9];
  const float* b_o = (const float*)d_in[10];
  float*       out = (float*)d_out;

  char* ws = (char*)d_ws;
  const size_t MB = 1u << 20;
  bf16_t* Qp  = (bf16_t*)(ws + 0 * MB);   // [4096,1024] bf16, 8 MB
  bf16_t* Kp  = (bf16_t*)(ws + 8 * MB);   // [4096,1024] bf16, 8 MB
  bf16_t* Vt  = (bf16_t*)(ws + 16 * MB);  // [B,H,Dk,S]  bf16, 8 MB
  bf16_t* Ob  = (bf16_t*)(ws + 24 * MB);  // [4096,1024] bf16, 8 MB
  bf16_t* wqb = (bf16_t*)(ws + 32 * MB);  // 2 MB each
  bf16_t* wkb = (bf16_t*)(ws + 34 * MB);
  bf16_t* wvb = (bf16_t*)(ws + 36 * MB);
  bf16_t* wob = (bf16_t*)(ws + 38 * MB);

  // 1) weight fp32 -> bf16 (once)
  {
    dim3 cb(256), cg((D_MODEL * D_MODEL) / (256 * 4));
    cvt_f32_to_bf16<<<cg, cb, 0, stream>>>(w_q, wqb);
    cvt_f32_to_bf16<<<cg, cb, 0, stream>>>(w_k, wkb);
    cvt_f32_to_bf16<<<cg, cb, 0, stream>>>(w_v, wvb);
    cvt_f32_to_bf16<<<cg, cb, 0, stream>>>(w_o, wob);
  }

  dim3 blk(32);
  dim3 gproj((BATCH * SEQ) / 16, D_MODEL / 64);  // (256, 16)

  // 2) Q/K/V projections (V stored head-transposed)
  proj_kernel<0><<<gproj, blk, 0, stream>>>(q, wqb, b_q, Qp);
  proj_kernel<0><<<gproj, blk, 0, stream>>>(k, wkb, b_k, Kp);
  proj_kernel<1><<<gproj, blk, 0, stream>>>(v, wvb, b_v, Vt);

  // 3) two-pass flash attention: grid (B*H, S/16)
  dim3 gattn(BATCH * NHEAD, SEQ / 16);
  attn_kernel<<<gattn, blk, 0, stream>>>(Qp, Kp, Vt, Ob);

  // 4) output projection -> fp32 d_out
  out_proj_kernel<<<gproj, blk, 0, stream>>>(Ob, wob, b_o, out);
}
